// UNetSplat_38663295598601
// MI455X (gfx1250) — matrix operand order repaired and checked by reading, not compiled, not run on hardware
//
#include <hip/hip_runtime.h>

// ---------------------------------------------------------------------------
// CDNA5 (gfx1250) UNet-splat forward.
//  * All 3x3 convs: bf16 implicit-GEMM on V_WMMA_F32_16X16X32_BF16 (wave32).
//  * LDS layouts match the WMMA VGPR layouts -> fragments load as ds_load_b128.
//  * Weight tiles staged global->LDS with GLOBAL_LOAD_ASYNC_TO_LDS_B128 when
//    the builtin exists (ASYNCcnt path), sync 16B vector copies otherwise.
//  * Activations kept in bf16 in workspace; weights packed fp32->bf16 per call.
//  * Decode head (softmax splat / upscale / compose) in fp32 VALU kernels.
// Param leaf order assumption: jax dict keys sorted ('ds','out','us'),
// each conv = (W, b) pair -> d_in[6 + 2*pair], d_in[7 + 2*pair].
// ---------------------------------------------------------------------------

typedef __bf16 bf16_t;
typedef __attribute__((ext_vector_type(16))) __bf16 v16bf;
typedef __attribute__((ext_vector_type(8)))  __bf16 v8bf;
typedef __attribute__((ext_vector_type(8)))  float  v8f;
typedef __attribute__((ext_vector_type(4)))  int    v4i;

#if defined(__gfx1250__) && __has_builtin(__builtin_amdgcn_global_load_async_to_lds_b128)
#define USE_ASYNC_LDS 1
// Builtin signature (from clang diagnostic): (v4i AS1*, v4i AS3*, imm, imm)
typedef __attribute__((address_space(1))) v4i* gv4i_p;   // global v4i*
typedef __attribute__((address_space(3))) v4i* lv4i_p;   // LDS v4i*
#else
#define USE_ASYNC_LDS 0
#endif

__device__ __forceinline__ void wait_async_all() {
#if USE_ASYNC_LDS
#if __has_builtin(__builtin_amdgcn_s_wait_asynccnt)
  __builtin_amdgcn_s_wait_asynccnt(0);
#else
  asm volatile("s_wait_asynccnt 0" ::: "memory");
#endif
#endif
}

// ------------------------------- tiny utils --------------------------------

__global__ void f32_to_bf16_k(const float* __restrict__ in, bf16_t* __restrict__ out,
                              long total) {
  long i = (long)blockIdx.x * blockDim.x + threadIdx.x;
  if (i >= total) return;
  out[i] = (bf16_t)in[i];
}

// Repack conv weight (OC,IC,3,3) fp32 -> bf16 [tap][OC][Kpad], K innermost,
// zero-padded K so the WMMA B-fragment (two 8-K runs) vector-loads.
__global__ void pack_w3_k(const float* __restrict__ w, bf16_t* __restrict__ dst,
                          int OC, int IC, int Kpad, long total) {
  long i = (long)blockIdx.x * blockDim.x + threadIdx.x;
  if (i >= total) return;
  int ci = (int)(i % Kpad);
  int oc = (int)((i / Kpad) % OC);
  int t  = (int)(i / ((long)Kpad * OC));
  float v = (ci < IC) ? w[((long)oc * IC + ci) * 9 + t] : 0.f;
  dst[i] = (bf16_t)v;
}

// ------------------------- WMMA 3x3 conv (bf16) ----------------------------
// Block = 256 threads = 8 waves. Tile: 64 pixels (x) x 32 out-channels.
// Wave (pg,og): pg in 0..3 picks 16-pixel group, og in 0..1 picks 16-oc group.
// Per K-chunk of 32 input channels: stage input slab (3rows x 66cols x 32cin)
// and weights (9 taps x 32n x 32k) into LDS, then 9 WMMAs (one per tap).
// A lane fragment = 2x ds_load_b128 (K=half*8+0..7 and 16+half*8+0..7).
// B lane fragment = 2x ds_load_b128 (K=half*8+0..7 and 16+half*8+0..7).
__global__ void __launch_bounds__(256)
conv3x3_wmma_k(const bf16_t* __restrict__ in, bf16_t* __restrict__ out,
               const bf16_t* __restrict__ wpk, const float* __restrict__ bias,
               int Cin, int Kpad, int Cout, int H, int W, int leaky) {
  constexpr int TW = 64;
  __shared__ bf16_t Ain[3][TW + 2][32];   // [row][col][cin_local]  ~12.7 KB
  __shared__ bf16_t Bw[9][32][32];        // [tap][n][k]            ~18.4 KB

  const int tid  = threadIdx.x;
  const int wv   = tid >> 5;
  const int lane = tid & 31;
  const int half = lane >> 4;
  const int lm   = lane & 15;
  const int pg   = wv & 3;
  const int og   = wv >> 2;

  const int y       = blockIdx.y;
  const int octiles = Cout >> 5;
  const int b       = blockIdx.z / octiles;
  const int n0      = (blockIdx.z % octiles) << 5;
  const int x0      = blockIdx.x * TW;

  const long inB = (long)b * Cin * H * W;
  v8f acc = {};

  for (int k0 = 0; k0 < Kpad; k0 += 32) {
    __syncthreads();
    // ---- stage B first: issue async global->LDS (or sync vector copies).
    for (int v = tid; v < 9 * 32 * 4; v += 256) {
      int g = v & 3;              // which 8-element group of k
      int n = (v >> 2) & 31;
      int t = v >> 7;
      const bf16_t* src = wpk + ((long)t * Cout + (n0 + n)) * Kpad + k0 + g * 8;
#if USE_ASYNC_LDS
      __builtin_amdgcn_global_load_async_to_lds_b128(
          (gv4i_p)(void*)src, (lv4i_p)(void*)&Bw[t][n][g * 8], 0, 0);
#else
      *(v8bf*)&Bw[t][n][g * 8] = *(const v8bf*)src;
#endif
      if (k0 + 32 < Kpad) __builtin_prefetch(src + 32, 0, 0);
    }
    // ---- stage A: input slab; consecutive tids walk x (coalesced global),
    //      LDS writes transpose to cin-innermost (overlaps with async B).
    for (int idx = tid; idx < 32 * 3 * (TW + 2); idx += 256) {
      int col = idx % (TW + 2);
      int r   = (idx / (TW + 2)) % 3;
      int c   = idx / (3 * (TW + 2));
      int xx  = x0 + col - 1;
      int yy  = y + r - 1;
      int ci  = k0 + c;
      float v = 0.f;
      if (ci < Cin && (unsigned)xx < (unsigned)W && (unsigned)yy < (unsigned)H)
        v = (float)in[inB + ((long)ci * H + yy) * W + xx];
      Ain[r][col][c] = (bf16_t)v;
    }
    wait_async_all();           // drain this wave's ASYNCcnt before barrier
    __syncthreads();

    #pragma unroll
    for (int t = 0; t < 9; ++t) {
      const int dy = t / 3, dx = t % 3;   // offset (-1) folded into staging
      // A fragment: lane lm = row (pixel), elements e<8 -> K=half*8+e,
      // e>=8 -> K=16+half*8+(e-8): two contiguous 8xbf16 runs.
      const bf16_t* ap = &Ain[dy][pg * 16 + lm + dx][half * 8];
      v8bf alo = *(const v8bf*)ap;
      v8bf ahi = *(const v8bf*)(ap + 16);
      v16bf av = __builtin_shufflevector(alo, ahi, 0, 1, 2, 3, 4, 5, 6, 7,
                                         8, 9, 10, 11, 12, 13, 14, 15);
      // B fragment: lane lm = column n, elements e<8 -> K=half*8+e,
      // e>=8 -> K=16+half*8+(e-8) (same order as A).
      const bf16_t* bp = &Bw[t][og * 16 + lm][half * 8];
      v8bf blo = *(const v8bf*)bp;
      v8bf bhi = *(const v8bf*)(bp + 16);
      v16bf bv = __builtin_shufflevector(blo, bhi, 0, 1, 2, 3, 4, 5, 6, 7,
                                         8, 9, 10, 11, 12, 13, 14, 15);
      acc = __builtin_amdgcn_wmma_f32_16x16x32_bf16(false, av, false, bv,
                                                    (short)0, acc, false, false);
    }
  }

  // C/D 16x16 f32 layout: VGPR r holds row m = r + half*8; n = lm.
  // A lane's 8 outputs are 8 consecutive x -> one packed b128 store when legal.
  const int oc  = n0 + og * 16 + lm;
  const float bz = bias[oc];
  v8bf ov;
  #pragma unroll
  for (int r = 0; r < 8; ++r) {
    float v = acc[r] + bz;
    if (leaky) v = (v >= 0.f) ? v : 0.3f * v;
    ov[r] = (bf16_t)v;
  }
  const int xbase = x0 + pg * 16 + half * 8;
  bf16_t* op = &out[(((long)b * Cout + oc) * H + y) * W + xbase];
  if (((W & 7) == 0) && (xbase + 8 <= W)) {
    *(v8bf*)op = ov;                        // 16B aligned: W%8==0, xbase%8==0
  } else {
    #pragma unroll
    for (int r = 0; r < 8; ++r)
      if (xbase + r < W) op[r] = ov[r];
  }
}

// ------------------------------ pool / resize ------------------------------

__global__ void maxpool2_k(const bf16_t* __restrict__ in, bf16_t* __restrict__ out,
                           int H, int W, long total) {
  long i = (long)blockIdx.x * blockDim.x + threadIdx.x;
  if (i >= total) return;
  int Wo = W >> 1, Ho = H >> 1;
  int xo = (int)(i % Wo);
  int yo = (int)((i / Wo) % Ho);
  long c = i / ((long)Wo * Ho);
  const bf16_t* p = in + (c * H + 2 * yo) * W + 2 * xo;
  float a = (float)p[0], b = (float)p[1], d = (float)p[W], e = (float)p[W + 1];
  out[i] = (bf16_t)fmaxf(fmaxf(a, b), fmaxf(d, e));
}

__global__ void avgpool2_k(const float* __restrict__ in, float* __restrict__ out,
                           int H, int W, long total) {
  long i = (long)blockIdx.x * blockDim.x + threadIdx.x;
  if (i >= total) return;
  int Wo = W >> 1, Ho = H >> 1;
  int xo = (int)(i % Wo);
  int yo = (int)((i / Wo) % Ho);
  long c = i / ((long)Wo * Ho);
  const float* p = in + (c * H + 2 * yo) * W + 2 * xo;
  out[i] = 0.25f * (p[0] + p[1] + p[W] + p[W + 1]);
}

// dst[B,Clow+Cskip,H,W] = concat(up2(low[B,Clow,H/2,W/2]), skip[B,Cskip,H,W])
__global__ void up2_concat_k(const bf16_t* __restrict__ low, const bf16_t* __restrict__ skip,
                             bf16_t* __restrict__ dst, int Clow, int Cskip,
                             int H, int W, long total) {
  long i = (long)blockIdx.x * blockDim.x + threadIdx.x;
  if (i >= total) return;
  int Ct = Clow + Cskip;
  int xx = (int)(i % W);
  int yy = (int)((i / W) % H);
  int c  = (int)((i / ((long)W * H)) % Ct);
  int b  = (int)(i / ((long)W * H * Ct));
  bf16_t v;
  if (c < Clow)
    v = low[(((long)b * Clow + c) * (H >> 1) + (yy >> 1)) * (W >> 1) + (xx >> 1)];
  else
    v = skip[(((long)b * Cskip + (c - Clow)) * H + yy) * W + xx];
  dst[i] = v;
}

// ------------------------------- decode head -------------------------------

__global__ void conv1x1_k(const bf16_t* __restrict__ in, const float* __restrict__ w,
                          const float* __restrict__ bias, float* __restrict__ out,
                          int IC, int OC, long HW, long total) {
  long i = (long)blockIdx.x * blockDim.x + threadIdx.x;
  if (i >= total) return;
  long p  = i % HW;
  int oc  = (int)((i / HW) % OC);
  int b   = (int)(i / (HW * OC));
  const bf16_t* ip = in + (long)b * IC * HW + p;
  const float*  wp = w + (long)oc * IC;
  float s = bias[oc];
  for (int c = 0; c < IC; ++c) s += (float)ip[(long)c * HW] * wp[c];
  out[i] = s;
}

// rendering = t_lambda*avg_prev + (1-t_lambda)*avg_color ; partition = w9*rendering
__global__ void render_partition_k(const float* __restrict__ wt, const float* __restrict__ pc,
                                   const float* __restrict__ cc, float* __restrict__ rend,
                                   float* __restrict__ part, int OC, long HW, long total) {
  long i = (long)blockIdx.x * blockDim.x + threadIdx.x;
  if (i >= total) return;
  long p = i % HW;
  int c  = (int)((i / HW) % 3);
  int b  = (int)(i / (HW * 3));
  const float* wb = wt + (long)b * OC * HW;
  float tl = wb[10 * HW + p];
  float w9 = wb[9 * HW + p];
  long ci = ((long)b * 3 + c) * HW + p;
  float r = tl * pc[ci] + (1.f - tl) * cc[ci];
  rend[i] = r;
  part[i] = w9 * r;
}

// Gather form of splat_unfold: out[y,x] = sum_k softmax_k(wt[.,ko+ :9, ys,xs]) * p[ys,xs]
__global__ void splat9_k(const float* __restrict__ wt, const float* __restrict__ p,
                         const float* __restrict__ addbuf, float* __restrict__ out,
                         int OC, int ko, int H, int W, long total) {
  long i = (long)blockIdx.x * blockDim.x + threadIdx.x;
  if (i >= total) return;
  long HW = (long)H * W;
  int xx = (int)(i % W);
  int yy = (int)((i / W) % H);
  int c  = (int)((i / HW) % 3);
  int b  = (int)(i / (HW * 3));
  const float* wb = wt + (long)b * OC * HW;
  const float* pb = p + ((long)b * 3 + c) * HW;
  float s = 0.f;
  for (int k = 0; k < 9; ++k) {
    int dy = k / 3 - 1, dx = k % 3 - 1;
    int ys = yy - dy, xs = xx - dx;
    if ((unsigned)ys < (unsigned)H && (unsigned)xs < (unsigned)W) {
      long sp = (long)ys * W + xs;
      float mx = -1e30f;
      for (int q = 0; q < 9; ++q) mx = fmaxf(mx, wb[(long)(ko + q) * HW + sp]);
      float dn = 0.f;
      for (int q = 0; q < 9; ++q) dn += __expf(wb[(long)(ko + q) * HW + sp] - mx);
      s += (__expf(wb[(long)(ko + k) * HW + sp] - mx) / dn) * pb[sp];
    }
  }
  if (addbuf) s += addbuf[i];
  out[i] = s;
}

// up2(den) * pixel_shuffle2(4*softmax(wt[:,11:15]))
__global__ void upscale_small_k(const float* __restrict__ den, const float* __restrict__ wt,
                                float* __restrict__ out, int OC, int H, int W, long total) {
  long i = (long)blockIdx.x * blockDim.x + threadIdx.x;
  if (i >= total) return;
  int W2 = W * 2, H2 = H * 2;
  long HW2 = (long)H2 * W2;
  int xx = (int)(i % W2);
  int yy = (int)((i / W2) % H2);
  int c  = (int)((i / HW2) % 3);
  int b  = (int)(i / (HW2 * 3));
  int xs = xx >> 1, ys = yy >> 1, q = (yy & 1) * 2 + (xx & 1);
  long HW = (long)H * W;
  long sp = (long)ys * W + xs;
  const float* wb = wt + (long)b * OC * HW;
  float mx = -1e30f;
  for (int t = 0; t < 4; ++t) mx = fmaxf(mx, wb[(long)(11 + t) * HW + sp]);
  float dn = 0.f;
  for (int t = 0; t < 4; ++t) dn += __expf(wb[(long)(11 + t) * HW + sp] - mx);
  float wv = 4.f * __expf(wb[(long)(11 + q) * HW + sp] - mx) / dn;
  out[i] = den[((long)b * 3 + c) * HW + sp] * wv;
}

// predict / color_ac / feature_ac
__global__ void final_compose_k(const float* __restrict__ wt, const float* __restrict__ den,
                                const float* __restrict__ prevs, const float* __restrict__ rend,
                                const float* __restrict__ pf, const float* __restrict__ ft,
                                float* __restrict__ predict, float* __restrict__ color_ac,
                                float* __restrict__ feat_ac, int OC, long HW, long total) {
  long i = (long)blockIdx.x * blockDim.x + threadIdx.x;
  if (i >= total) return;
  long p = i % HW;
  int c  = (int)((i / HW) % 8);
  int b  = (int)(i / (HW * 8));
  const float* wb = wt + (long)b * OC * HW;
  float tl = wb[10 * HW + p];
  if (c < 3) {
    float z = wb[11 * HW + p];
    float tmu = 1.f / (1.f + __expf(-z));
    long ci = ((long)b * 3 + c) * HW + p;
    predict[ci]  = tmu * prevs[ci] + (1.f - tmu) * den[ci];
    color_ac[ci] = rend[ci];
  }
  long fi = ((long)b * 8 + c) * HW + p;
  feat_ac[fi] = tl * pf[fi] + (1.f - tl) * ft[fi];
}

// --------------------------------- driver ----------------------------------

extern "C" void kernel_launch(void* const* d_in, const int* in_sizes, int n_in,
                              void* d_out, int out_size, void* d_ws, size_t ws_size,
                              hipStream_t stream) {
  (void)in_sizes; (void)n_in; (void)out_size; (void)ws_size;
  const int B = 2;

  const float* x_in         = (const float*)d_in[0];
  const float* prev_color   = (const float*)d_in[1];
  const float* color        = (const float*)d_in[2];
  const float* previous     = (const float*)d_in[3];
  const float* prev_feature = (const float*)d_in[4];
  const float* feature      = (const float*)d_in[5];

  // Conv (W,b) pair index -> d_in leaf. Pairs in order:
  // ds: 0:(64,12) 1:(64,64) 2:(96,64) 3:(128,96) 4:(192,128) 5:(256,192)
  //     6:(384,256) 7:(512,384) 8:(512,512) 9:(384,512)
  // out: 10:(21,64) 11:(15,96) 12:(15,128) 13:(15,192) 14:(15,256)
  // us: 15:(384,768) 16:(384,384) 17:(256,384) 18:(256,448) 19:(192,256)
  //     20:(192,320) 21:(128,192) 22:(128,224) 23:(96,128) 24:(96,160) 25:(64,96) 26:(64,64)
  auto Wp = [&](int pair) { return (const float*)d_in[6 + 2 * pair]; };
  auto Bp = [&](int pair) { return (const float*)d_in[6 + 2 * pair + 1]; };

  char* ws = (char*)d_ws;
  size_t cur = 0;
  auto alloc = [&](size_t bytes) -> void* {
    cur = (cur + 255) & ~(size_t)255;
    void* p = ws + cur;
    cur += bytes;
    return p;
  };
  auto nb = [](long t) { return (unsigned)((t + 255) / 256); };
  auto abf = [&](int C, int H) {
    return (bf16_t*)alloc((size_t)B * C * H * H * sizeof(bf16_t));
  };
  auto af32 = [&](int C, int H) {
    return (float*)alloc((size_t)B * C * H * H * sizeof(float));
  };

  auto conv3 = [&](const bf16_t* in, bf16_t* out, int pair, int Cin, int Cout, int H) {
    int Kpad = (Cin + 31) & ~31;
    bf16_t* wpk = (bf16_t*)alloc((size_t)9 * Kpad * Cout * sizeof(bf16_t));
    long tp = (long)9 * Kpad * Cout;
    pack_w3_k<<<nb(tp), 256, 0, stream>>>(Wp(pair), wpk, Cout, Cin, Kpad, tp);
    dim3 grid((H + 63) / 64, H, B * (Cout >> 5));
    conv3x3_wmma_k<<<grid, 256, 0, stream>>>(in, out, wpk, Bp(pair),
                                             Cin, Kpad, Cout, H, H, 1);
  };
  auto pool = [&](const bf16_t* in, bf16_t* out, int C, int H) {
    long t = (long)B * C * (H / 2) * (H / 2);
    maxpool2_k<<<nb(t), 256, 0, stream>>>(in, out, H, H, t);
  };
  auto upcat = [&](const bf16_t* low, const bf16_t* skip, bf16_t* dst,
                   int Clow, int Cskip, int H) {
    long t = (long)B * (Clow + Cskip) * H * H;
    up2_concat_k<<<nb(t), 256, 0, stream>>>(low, skip, dst, Clow, Cskip, H, H, t);
  };

  // ---------------- encoder ----------------
  bf16_t* xbf = abf(12, 384);
  { long t = (long)B * 12 * 384 * 384;
    f32_to_bf16_k<<<nb(t), 256, 0, stream>>>(x_in, xbf, t); }

  bf16_t* t0  = abf(64, 384);  conv3(xbf, t0, 0, 12, 64, 384);
  bf16_t* s0  = abf(64, 384);  conv3(t0, s0, 1, 64, 64, 384);
  bf16_t* p0  = abf(64, 192);  pool(s0, p0, 64, 384);
  bf16_t* s1  = abf(96, 192);  conv3(p0, s1, 2, 64, 96, 192);
  bf16_t* p1  = abf(96, 96);   pool(s1, p1, 96, 192);
  bf16_t* s2  = abf(128, 96);  conv3(p1, s2, 3, 96, 128, 96);
  bf16_t* p2  = abf(128, 48);  pool(s2, p2, 128, 96);
  bf16_t* s3  = abf(192, 48);  conv3(p2, s3, 4, 128, 192, 48);
  bf16_t* p3  = abf(192, 24);  pool(s3, p3, 192, 48);
  bf16_t* e4a = abf(256, 24);  conv3(p3, e4a, 5, 192, 256, 24);
  bf16_t* s4  = abf(384, 24);  conv3(e4a, s4, 6, 256, 384, 24);
  bf16_t* p4  = abf(384, 12);  pool(s4, p4, 384, 24);
  bf16_t* e5a = abf(512, 12);  conv3(p4, e5a, 7, 384, 512, 12);
  bf16_t* e5b = abf(512, 12);  conv3(e5a, e5b, 8, 512, 512, 12);
  bf16_t* s5  = abf(384, 12);  conv3(e5b, s5, 9, 512, 384, 12);

  // ---------------- color pyramids (avgpool 2^j) ----------------
  const float* pcp[5]; const float* ccp[5];
  pcp[0] = prev_color; ccp[0] = color;
  {
    int hh = 384;
    for (int j = 1; j < 5; ++j) {
      int ho = hh >> 1;
      float* a  = af32(3, ho);
      float* c2 = af32(3, ho);
      long t = (long)B * 3 * ho * ho;
      avgpool2_k<<<nb(t), 256, 0, stream>>>(pcp[j - 1], a, hh, hh, t);
      avgpool2_k<<<nb(t), 256, 0, stream>>>(ccp[j - 1], c2, hh, hh, t);
      pcp[j] = a; ccp[j] = c2; hh = ho;
    }
  }

  // ---------------- decode helper ----------------
  auto decode = [&](const bf16_t* feat, int Cf, int outPair, int OC, int Hj,
                    const float* avgp, const float* avgc, const float* addus,
                    float** wt_o, float** rend_o, float** den_o) {
    long HWj = (long)Hj * Hj;
    float* wt = af32(OC, Hj);
    long t1 = (long)B * OC * HWj;
    conv1x1_k<<<nb(t1), 256, 0, stream>>>(feat, Wp(outPair), Bp(outPair), wt,
                                          Cf, OC, HWj, t1);
    float* rend = af32(3, Hj);
    float* part = af32(3, Hj);
    long t2 = (long)B * 3 * HWj;
    render_partition_k<<<nb(t2), 256, 0, stream>>>(wt, avgp, avgc, rend, part,
                                                   OC, HWj, t2);
    float* den = af32(3, Hj);
    splat9_k<<<nb(t2), 256, 0, stream>>>(wt, part, addus, den, OC, 0, Hj, Hj, t2);
    *wt_o = wt; *rend_o = rend; *den_o = den;
  };

  // ---------------- decoder level i=0 (j=4, 24x24) ----------------
  bf16_t* cat4 = abf(768, 24); upcat(s5, s4, cat4, 384, 384, 24);
  bf16_t* d4a  = abf(384, 24); conv3(cat4, d4a, 15, 768, 384, 24);
  bf16_t* d4b  = abf(384, 24); conv3(d4a, d4b, 16, 384, 384, 24);
  bf16_t* d4c  = abf(256, 24); conv3(d4b, d4c, 17, 384, 256, 24);
  float *wt4, *rend4, *den4;
  decode(d4c, 256, 14, 15, 24, pcp[4], ccp[4], nullptr, &wt4, &rend4, &den4);
  float* us43 = af32(3, 48);
  { long t = (long)B * 3 * 48 * 48;
    upscale_small_k<<<nb(t), 256, 0, stream>>>(den4, wt4, us43, 15, 24, 24, t); }

  // ---------------- i=1 (j=3, 48x48) ----------------
  bf16_t* cat3 = abf(448, 48); upcat(d4c, s3, cat3, 256, 192, 48);
  bf16_t* d3a  = abf(256, 48); conv3(cat3, d3a, 18, 448, 256, 48);
  bf16_t* d3b  = abf(192, 48); conv3(d3a, d3b, 19, 256, 192, 48);
  float *wt3, *rend3, *den3;
  decode(d3b, 192, 13, 15, 48, pcp[3], ccp[3], us43, &wt3, &rend3, &den3);
  float* us32 = af32(3, 96);
  { long t = (long)B * 3 * 96 * 96;
    upscale_small_k<<<nb(t), 256, 0, stream>>>(den3, wt3, us32, 15, 48, 48, t); }

  // ---------------- i=2 (j=2, 96x96) ----------------
  bf16_t* cat2 = abf(320, 96); upcat(d3b, s2, cat2, 192, 128, 96);
  bf16_t* d2a  = abf(192, 96); conv3(cat2, d2a, 20, 320, 192, 96);
  bf16_t* d2b  = abf(128, 96); conv3(d2a, d2b, 21, 192, 128, 96);
  float *wt2, *rend2, *den2;
  decode(d2b, 128, 12, 15, 96, pcp[2], ccp[2], us32, &wt2, &rend2, &den2);
  float* us21 = af32(3, 192);
  { long t = (long)B * 3 * 192 * 192;
    upscale_small_k<<<nb(t), 256, 0, stream>>>(den2, wt2, us21, 15, 96, 96, t); }

  // ---------------- i=3 (j=1, 192x192) ----------------
  bf16_t* cat1 = abf(224, 192); upcat(d2b, s1, cat1, 128, 96, 192);
  bf16_t* d1a  = abf(128, 192); conv3(cat1, d1a, 22, 224, 128, 192);
  bf16_t* d1b  = abf(96, 192);  conv3(d1a, d1b, 23, 128, 96, 192);
  float *wt1, *rend1, *den1;
  decode(d1b, 96, 11, 15, 192, pcp[1], ccp[1], us21, &wt1, &rend1, &den1);
  float* us10 = af32(3, 384);
  { long t = (long)B * 3 * 384 * 384;
    upscale_small_k<<<nb(t), 256, 0, stream>>>(den1, wt1, us10, 15, 192, 192, t); }

  // ---------------- i=4 (j=0, 384x384, final) ----------------
  bf16_t* cat0 = abf(160, 384); upcat(d1b, s0, cat0, 96, 64, 384);
  bf16_t* d0a  = abf(96, 384);  conv3(cat0, d0a, 24, 160, 96, 384);
  bf16_t* d0b  = abf(64, 384);  conv3(d0a, d0b, 25, 96, 64, 384);
  bf16_t* d0c  = abf(64, 384);  conv3(d0b, d0c, 26, 64, 64, 384);
  float *wt0, *rend0, *den0;
  decode(d0c, 64, 10, 21, 384, pcp[0], ccp[0], us10, &wt0, &rend0, &den0);

  float* prevs = af32(3, 384);
  { long t = (long)B * 3 * 384 * 384;
    splat9_k<<<nb(t), 256, 0, stream>>>(wt0, previous, nullptr, prevs, 21, 12,
                                        384, 384, t); }

  long HWf = (long)384 * 384;
  float* predict  = (float*)d_out;
  float* color_ac = predict + (long)B * 3 * HWf;
  float* feat_ac  = color_ac + (long)B * 3 * HWf;
  { long t = (long)B * 8 * HWf;
    final_compose_k<<<nb(t), 256, 0, stream>>>(wt0, den0, prevs, rend0,
                                               prev_feature, feature,
                                               predict, color_ac, feat_ac,
                                               21, HWf, t); }
  (void)rend4; (void)rend3; (void)rend2; (void)rend1;
}
